// ScannedRNN_59992103191036
// MI455X (gfx1250) — matrix operand order repaired and checked by reading, not compiled, not run on hardware
//
#include <hip/hip_runtime.h>

#define T_LEN 512
#define BATCH 256
#define HID   256
#define INP   256
#define NWG   16                 // persistent workgroups; each owns HID/NWG h-columns
#define WG_COLS (HID / NWG)      // 16

typedef __bf16 bf16;
typedef __bf16 v16bf __attribute__((ext_vector_type(16)));
typedef __bf16 v8bf  __attribute__((ext_vector_type(8)));
typedef float  v8f   __attribute__((ext_vector_type(8)));
typedef float  v4f   __attribute__((ext_vector_type(4)));

__device__ __forceinline__ float sigmoidf_(float x) {
  return 1.0f / (1.0f + __expf(-x));
}

// Reset grid-barrier counter and seed h0 = carry (bf16).
__global__ void __launch_bounds__(256) gru_init(const float* __restrict__ carry,
                                                bf16* __restrict__ hbuf0,
                                                unsigned* __restrict__ counter) {
  int i = blockIdx.x * blockDim.x + threadIdx.x;
  if (i == 0) *counter = 0u;
  if (i < BATCH * HID) hbuf0[i] = (bf16)carry[i];
}

// Parallel pre-pass: convert seq_inputs (T,B,INP) fp32 -> bf16.
// Removes all v_cvt_pk_bf16_f32 work from the 512-step serial recurrence.
__global__ void __launch_bounds__(256) xconv(const float* __restrict__ x,
                                             bf16* __restrict__ xbf) {
  size_t i = ((size_t)blockIdx.x * blockDim.x + threadIdx.x) * 8;
  v4f a = *(const v4f*)(x + i);
  v4f b = *(const v4f*)(x + i + 4);
  v8bf o;
  #pragma unroll
  for (int j = 0; j < 4; ++j) { o[j] = (bf16)a[j]; o[4 + j] = (bf16)b[j]; }
  *(v8bf*)(xbf + i) = o;
}

// Persistent fused GRU scan. Each WG owns columns [wg*16, wg*16+16) of h for
// all three gates; wave w owns batch rows [w*32, w*32+32).
// XBF=true: x pre-converted to bf16 in workspace; XBF=false: read fp32 x, cvt in-loop.
template <bool XBF>
__global__ void __launch_bounds__(256) gru_scan(
    const float* __restrict__ x,      // (T, B, INP) fp32
    const bf16*  __restrict__ xbf,    // (T, B, INP) bf16 (valid iff XBF)
    const int*   __restrict__ rst,    // (T, B) int32
    const float* __restrict__ w_ih,   // (3H, INP)
    const float* __restrict__ w_hh,   // (3H, HID)
    const float* __restrict__ b_ih,   // (3H)
    const float* __restrict__ b_hh,   // (3H)
    float* __restrict__ out,          // [B*H h_final][T*B*H outputs]
    bf16* __restrict__ hb0,
    bf16* __restrict__ hb1,
    unsigned* __restrict__ counter)
{
  // Weight fragments pre-swizzled to WMMA B-operand layout:
  // [gate][kchunk][lane][slot], slot s -> K = kchunk*32 + s + (s&8) + 8*laneHi, N = lane&15.
  __shared__ bf16 wfrag[3 * 16 * 32 * 16];   // 48 KB

  const int tid  = threadIdx.x;
  const int wave = tid >> 5;
  const int lane = tid & 31;
  const int hi   = (lane >> 4) & 1;
  const int ln   = lane & 15;
  const int wg   = blockIdx.x;
  const int col  = wg * WG_COLS + ln;        // N index / h column for this lane

  // ---- one-time: build swizzled weights in LDS (W_cat = [w_ih | w_hh]) ----
  for (int i = 0; i < 6; ++i) {              // 48 (gate,kchunk) combos / 8 waves
    int c = wave * 6 + i;
    int g = c / 16, q = c % 16;
    int row = g * HID + wg * WG_COLS + ln;   // W_cat row = gate output column
    bf16* dst = &wfrag[((g * 16 + q) * 32 + lane) * 16];
    int kb = q * 32 + hi * 8;
    #pragma unroll
    for (int s = 0; s < 16; ++s) {
      int k = kb + s + (s & 8);
      float w = (k < INP) ? w_ih[row * INP + k] : w_hh[row * HID + (k - INP)];
      dst[s] = (bf16)w;
    }
  }

  // per-lane gate biases (N depends only on lane&15)
  const float brC  = b_ih[col]           + b_hh[col];
  const float bzC  = b_ih[HID + col]     + b_hh[HID + col];
  const float binC = b_ih[2 * HID + col];          // kept separate: n-gate needs
  const float bhnC = b_hh[2 * HID + col];          // i_n + r*(h_n) structure

  __syncthreads();

  const int b0 = wave * 32;

  for (int t = 0; t < T_LEN; ++t) {
    const bf16* hcur = (t & 1) ? hb1 : hb0;
    bf16*       hnxt = (t & 1) ? hb0 : hb1;
    const float* xt   = x   + (size_t)t * BATCH * INP;
    const bf16*  xtb  = xbf + (size_t)t * BATCH * INP;
    const int*   rt   = rst + (size_t)t * BATCH;

    if (t + 1 < T_LEN) {   // gfx1250 global_prefetch for next step's activations
      if (XBF)
        __builtin_prefetch(xbf + (size_t)(t + 1) * BATCH * INP + (size_t)(b0 + ln) * INP, 0, 1);
      else
        __builtin_prefetch(x + (size_t)(t + 1) * BATCH * INP + (size_t)(b0 + ln) * INP, 0, 1);
    }

    #pragma unroll
    for (int mt = 0; mt < 2; ++mt) {
      const int rb   = b0 + mt * 16;
      const int rowA = rb + ln;              // A-operand row (lanes 0-15 & 16-31 share M)
      const int rstA = rt[rowA];

      v8f accR = {}, accZ = {}, accIN = {}, accHN = {};

      #pragma unroll
      for (int q = 0; q < 16; ++q) {
        // ---- A fragment (16x32 bf16): slot s -> K = q*32 + s + (s&8) + 8*hi ----
        v16bf a;
        if (q < 8) {                         // K in x-range
          if (XBF) {                         // bf16 direct: 2x16B loads, no cvt
            const bf16* xr = xtb + (size_t)rowA * INP + q * 32 + hi * 8;
            v8bf lo = *(const v8bf*)(xr);
            v8bf hv = *(const v8bf*)(xr + 16);
            #pragma unroll
            for (int j = 0; j < 8; ++j) { a[j] = lo[j]; a[8 + j] = hv[j]; }
          } else {                           // fp32 fallback: load + cvt in-loop
            const float* xr = xt + (size_t)rowA * INP + q * 32 + hi * 8;
            v4f lo0 = *(const v4f*)(xr);
            v4f lo1 = *(const v4f*)(xr + 4);
            v4f hi0 = *(const v4f*)(xr + 16);
            v4f hi1 = *(const v4f*)(xr + 20);
            #pragma unroll
            for (int j = 0; j < 4; ++j) {
              a[j]      = (bf16)lo0[j];
              a[4 + j]  = (bf16)lo1[j];
              a[8 + j]  = (bf16)hi0[j];
              a[12 + j] = (bf16)hi1[j];
            }
          }
        } else {                             // K in h-range: bf16, masked by reset
          v8bf lo = {}, hv = {};
          if (!rstA) {
            const bf16* hr = hcur + (size_t)rowA * HID + (q - 8) * 32 + hi * 8;
            lo = *(const v8bf*)(hr);
            hv = *(const v8bf*)(hr + 16);
          }
          #pragma unroll
          for (int j = 0; j < 8; ++j) { a[j] = lo[j]; a[8 + j] = hv[j]; }
        }

        // ---- B fragments straight from pre-swizzled LDS (contiguous 32B/lane) ----
        const v16bf* wf = (const v16bf*)wfrag;
        v16bf bR = wf[(0 * 16 + q) * 32 + lane];
        v16bf bZ = wf[(1 * 16 + q) * 32 + lane];
        v16bf bN = wf[(2 * 16 + q) * 32 + lane];

        accR = __builtin_amdgcn_wmma_f32_16x16x32_bf16(false, a, false, bR, (short)0, accR, false, false);
        accZ = __builtin_amdgcn_wmma_f32_16x16x32_bf16(false, a, false, bZ, (short)0, accZ, false, false);
        if (q < 8)
          accIN = __builtin_amdgcn_wmma_f32_16x16x32_bf16(false, a, false, bN, (short)0, accIN, false, false);
        else
          accHN = __builtin_amdgcn_wmma_f32_16x16x32_bf16(false, a, false, bN, (short)0, accHN, false, false);
      }

      // ---- GRU elementwise, fully in registers (C layout: VGPR e -> M = e + 8*hi) ----
      #pragma unroll
      for (int e = 0; e < 8; ++e) {
        const int row = rb + e + hi * 8;
        float r = sigmoidf_(accR[e] + brC);
        float z = sigmoidf_(accZ[e] + bzC);
        float n = tanhf(accIN[e] + binC + r * (accHN[e] + bhnC));
        float hu = rt[row] ? 0.0f : (float)hcur[(size_t)row * HID + col];
        float hnew = (1.0f - z) * n + z * hu;
        out[(size_t)BATCH * HID + ((size_t)t * BATCH + row) * HID + col] = hnew;
        hnxt[(size_t)row * HID + col] = (bf16)hnew;
        if (t == T_LEN - 1) out[(size_t)row * HID + col] = hnew;   // h_final
      }
    }

    // ---- grid-wide barrier: publish hnxt, 16 arrivals/step ----
    __threadfence();
    __syncthreads();
    if (tid == 0) {
      __hip_atomic_fetch_add(counter, 1u, __ATOMIC_RELEASE, __HIP_MEMORY_SCOPE_AGENT);
      const unsigned target = (unsigned)(NWG * (t + 1));
      while (__hip_atomic_load(counter, __ATOMIC_ACQUIRE, __HIP_MEMORY_SCOPE_AGENT) < target) {
        __builtin_amdgcn_s_sleep(1);
      }
    }
    __syncthreads();
    __threadfence();
  }
}

extern "C" void kernel_launch(void* const* d_in, const int* in_sizes, int n_in,
                              void* d_out, int out_size, void* d_ws, size_t ws_size,
                              hipStream_t stream) {
  const float* carry = (const float*)d_in[0];
  const float* x     = (const float*)d_in[1];
  const int*   rst   = (const int*)d_in[2];
  const float* w_ih  = (const float*)d_in[3];
  const float* w_hh  = (const float*)d_in[4];
  const float* b_ih  = (const float*)d_in[5];
  const float* b_hh  = (const float*)d_in[6];
  float* out = (float*)d_out;

  // workspace layout:
  //   [counter, 256B pad][hbuf0 bf16 B*H][hbuf1 bf16 B*H][xbf bf16 T*B*INP (optional)]
  char* ws = (char*)d_ws;
  unsigned* counter = (unsigned*)ws;
  bf16* hb0 = (bf16*)(ws + 256);
  bf16* hb1 = (bf16*)(ws + 256 + (size_t)BATCH * HID * sizeof(bf16));
  bf16* xbf = (bf16*)(ws + 256 + 2 * (size_t)BATCH * HID * sizeof(bf16));

  const size_t need_xbf = 256 + 2 * (size_t)BATCH * HID * sizeof(bf16)
                        + (size_t)T_LEN * BATCH * INP * sizeof(bf16);
  const bool use_xbf = (ws_size >= need_xbf);   // ws_size fixed -> deterministic

  gru_init<<<(BATCH * HID + 255) / 256, 256, 0, stream>>>(carry, hb0, counter);

  if (use_xbf) {
    const size_t total = (size_t)T_LEN * BATCH * INP;        // 33.5M elems
    xconv<<<(unsigned)(total / (256 * 8)), 256, 0, stream>>>(x, xbf);
    gru_scan<true><<<NWG, 256, 0, stream>>>(x, xbf, rst, w_ih, w_hh, b_ih, b_hh,
                                            out, hb0, hb1, counter);
  } else {
    gru_scan<false><<<NWG, 256, 0, stream>>>(x, xbf, rst, w_ih, w_hh, b_ih, b_hh,
                                             out, hb0, hb1, counter);
  }
}
